// LlamaCrossAttention_73847667687794
// MI455X (gfx1250) — matrix operand order repaired and checked by reading, not compiled, not run on hardware
//
#include <hip/hip_runtime.h>

#define BATCH 2
#define LQ    2048
#define LKV   576
#define HID   4096
#define NH    32
#define DH    128

typedef __attribute__((ext_vector_type(16))) __bf16 v16bf;
typedef __attribute__((ext_vector_type(8)))  __bf16 v8bf;
typedef __attribute__((ext_vector_type(4)))  __bf16 v4bf;
typedef __attribute__((ext_vector_type(8)))  float  v8f;
typedef __attribute__((ext_vector_type(4)))  int    v4i;

union V16U { v16bf v; v8bf h[2]; };

__device__ __forceinline__ v16bf cat16(v8bf lo, v8bf hi) {
  V16U u; u.h[0] = lo; u.h[1] = hi; return u.v;
}

__device__ __forceinline__ __bf16 to_bf16(float f) {
  unsigned u = __builtin_bit_cast(unsigned, f);
  u += 0x7fffu + ((u >> 16) & 1u);          // round-to-nearest-even
  unsigned short s = (unsigned short)(u >> 16);
  return __builtin_bit_cast(__bf16, s);
}

// ---------------------------------------------------------------------------
// CDNA5 async global->LDS copy (ASYNCcnt-tracked) + wait.
// Builtin prototype (from hipcc diagnostic): (v4i AS1*, v4i AS3*, Ii, Ii).
// Address-space-changing casts are routed through integers (flat global value
// == AS1 address; flat LDS addr low 32 bits == LDS offset per ISA aperture).
// ---------------------------------------------------------------------------
__device__ __forceinline__ void async_copy_b128(void* lds_dst, const void* gsrc) {
#if __has_builtin(__builtin_amdgcn_global_load_async_to_lds_b128)
  typedef __attribute__((address_space(1))) v4i g_v4i;
  typedef __attribute__((address_space(3))) v4i l_v4i;
  __builtin_amdgcn_global_load_async_to_lds_b128(
      (g_v4i*)(unsigned long long)gsrc,
      (l_v4i*)(unsigned)(unsigned long long)lds_dst, 0, 0);
#else
  unsigned lds_off = (unsigned)(unsigned long long)lds_dst;
  asm volatile("global_load_async_to_lds_b128 %0, %1, off"
               :: "v"(lds_off), "v"(gsrc) : "memory");
#endif
}

template <int N>
__device__ __forceinline__ void wait_async() {
#if __has_builtin(__builtin_amdgcn_s_wait_asynccnt)
  __builtin_amdgcn_s_wait_asynccnt(N);
#else
  asm volatile("s_wait_asynccnt %0" :: "i"(N) : "memory");
#endif
}

// ---------------------------------------------------------------------------
// Elementwise f32 -> bf16 conversion (vectorized: float4 load, 8B store)
// ---------------------------------------------------------------------------
__global__ __launch_bounds__(256) void cvt_f32_bf16(const float* __restrict__ src,
                                                    __bf16* __restrict__ dst,
                                                    long long n) {
  long long i = ((long long)blockIdx.x * blockDim.x + threadIdx.x) * 4;
  if (i + 3 < n) {
    float4 f = *(const float4*)(src + i);
    v4bf o;
    o[0] = to_bf16(f.x); o[1] = to_bf16(f.y);
    o[2] = to_bf16(f.z); o[3] = to_bf16(f.w);
    *(v4bf*)(dst + i) = o;
  }
}

// ---------------------------------------------------------------------------
// C[M,N] = scale * (A[M,K] @ W[N,K]^T)      (torch Linear, no bias)
// WG = 256 threads = 8 waves; WG tile 128(M) x 128(N), each wave 16(M)x128(N).
// W tile (128 rows x 32 k, bf16) staged in double-buffered LDS via
// global_load_async_to_lds_b128 (shared by all 8 waves); A streamed from
// global with one-step register prefetch. Rows padded to 40 elems (80B) so
// ds_load_b128 fragment reads are bank-conflict-free (stride 20 words).
// All 8 B-fragments are gathered into registers first so the 8 WMMAs issue
// back-to-back behind a single dscnt wait.
// mode 0: out bf16 row-major [M,N]
// mode 1: out bf16 transposed-per-batch: dst[((m/Lrows)*N + n)*Lrows + m%Lrows]
// mode 2: out f32 row-major [M,N]
// ---------------------------------------------------------------------------
#define BROW 40   // padded row length (elements) in LDS

__global__ __launch_bounds__(256) void gemm_bf16_nt(const __bf16* __restrict__ A,
                                                    const __bf16* __restrict__ W,
                                                    void* __restrict__ out,
                                                    int M, int N, int K,
                                                    int mode, float scale, int Lrows) {
  __shared__ __bf16 Bsm[2][128 * BROW];   // 2 x 10 KB

  const int tid  = threadIdx.x;
  const int lane = tid & 31;
  const int wave = tid >> 5;
  const int ln = lane & 15;
  const int hi = lane >> 4;
  const int n0 = blockIdx.x * 128;
  const int mwave = blockIdx.y * 128 + wave * 16;

  // staging assignment: 128 rows x 64B; each thread copies 32B (2 async b128)
  const int srow  = tid >> 1;      // 0..127
  const int shalf = tid & 1;       // 0/1 -> element offset shalf*16
  const __bf16* Wstage = W + (size_t)(n0 + srow) * K + shalf * 16;
  __bf16* Lstage = &Bsm[0][0] + srow * BROW + shalf * 16;

  // A-fragment: row m = mwave+ln, K chunks {hi*8..} and {16+hi*8..}
  const __bf16* Arow = A + (size_t)(mwave + ln) * K + hi * 8;

  const int S = K / 32;

  // prologue: stage k-step 0, prefetch A k-step 0
  async_copy_b128(Lstage, Wstage);
  async_copy_b128(Lstage + 8, Wstage + 8);
  v8bf a_lo = *(const v8bf*)(Arow);
  v8bf a_hi = *(const v8bf*)(Arow + 16);

  v8f acc[8] = {};
  for (int s = 0; s < S; ++s) {
    const int buf = s & 1;
    v8bf a_lo_n, a_hi_n;
    if (s + 1 < S) {
      // issue next-buffer async copies + next A prefetch, then wait for the
      // CURRENT buffer (<=2 leaves the 2 new copies in flight)
      const __bf16* wn = Wstage + (size_t)(s + 1) * 32;
      __bf16* lnxt = Lstage + (buf ^ 1) * (128 * BROW) - buf * (128 * BROW);
      async_copy_b128(lnxt, wn);
      async_copy_b128(lnxt + 8, wn + 8);
      a_lo_n = *(const v8bf*)(Arow + (s + 1) * 32);
      a_hi_n = *(const v8bf*)(Arow + (s + 1) * 32 + 16);
      wait_async<2>();
    } else {
      wait_async<0>();
    }
    __syncthreads();   // whole W tile visible to all waves

    v16bf a = cat16(a_lo, a_hi);
    const __bf16* Bbase = &Bsm[buf][0] + ln * BROW + hi * 8;

    // gather all 8 B-fragments first (one ds clause, one wait) ...
    v16bf bfrag[8];
#pragma unroll
    for (int j = 0; j < 8; ++j) {
      const __bf16* bp = Bbase + j * 16 * BROW;
      bfrag[j] = cat16(*(const v8bf*)bp, *(const v8bf*)(bp + 16));
    }
    // ... then fire the WMMA chain back-to-back (no inter-WMMA hazards)
#pragma unroll
    for (int j = 0; j < 8; ++j)
      acc[j] = __builtin_amdgcn_wmma_f32_16x16x32_bf16(false, a, false, bfrag[j],
                                                       (short)0, acc[j], false, false);

    a_lo = a_lo_n; a_hi = a_hi_n;
    __syncthreads();   // all waves done reading buf before it is overwritten
  }

  // C layout: VGPR r -> M = mwave + r + hi*8, N = n0 + j*16 + ln
  const int mbase = mwave + hi * 8;
#pragma unroll
  for (int j = 0; j < 8; ++j) {
    const int n = n0 + j * 16 + ln;
#pragma unroll
    for (int r = 0; r < 8; ++r) {
      const int m = mbase + r;
      float v = acc[j][r] * scale;
      if (mode == 2) {
        ((float*)out)[(size_t)m * N + n] = v;
      } else if (mode == 1) {
        int bbi = m / Lrows, rr = m - bbi * Lrows;
        ((__bf16*)out)[((size_t)bbi * N + n) * (size_t)Lrows + rr] = to_bf16(v);
      } else {
        ((__bf16*)out)[(size_t)m * N + n] = to_bf16(v);
      }
    }
  }
}

// ---------------------------------------------------------------------------
// Attention core: S = Qs Kh^T + mask ; P = softmax(S) ; O = P Vh
//   Qb  : bf16 [B*LQ, HID]   (already scaled by 1/sqrt(D))
//   Kb  : bf16 [B*LKV, HID]
//   Vt  : bf16 [B, HID, LKV] (V transposed per head: [d, kv] contiguous in kv)
//   mask: f32  [B,1,LQ,LKV]
//   Attn: bf16 [B*LQ, HID]
// 128 threads = 4 waves; each wave owns 16 q rows. Full-row softmax over 576,
// scores staged in LDS f32, P re-staged as bf16 in WMMA A-operand order.
// ---------------------------------------------------------------------------
__global__ __launch_bounds__(128) void attn_fwd(const __bf16* __restrict__ Qb,
                                                const __bf16* __restrict__ Kb,
                                                const __bf16* __restrict__ Vt,
                                                const float* __restrict__ mask,
                                                __bf16* __restrict__ Attn) {
  extern __shared__ char smem[];
  const int lane = threadIdx.x & 31;
  const int wave = threadIdx.x >> 5;      // 0..3
  const int ln = lane & 15;
  const int hi = lane >> 4;
  const int h = blockIdx.y;
  const int b = blockIdx.z;
  const int q0 = blockIdx.x * 64 + wave * 16;

  float*  Sbuf = (float*)(smem + wave * (16 * LKV * 4));
  __bf16* Pbuf = (__bf16*)(smem + 4 * (16 * LKV * 4) + wave * (16 * LKV * 2));
  float*  sums = (float*)(smem + 4 * (16 * LKV * 4) + 4 * (16 * LKV * 2) + wave * 16 * 4);

  // ---- preload 4 Q A-fragments (16 x 128 over 4 K-chunks of 32) ----
  const __bf16* Qrow = Qb + (size_t)(b * LQ + q0 + ln) * HID + h * DH + hi * 8;
  v16bf qa[4];
#pragma unroll
  for (int c = 0; c < 4; ++c)
    qa[c] = cat16(*(const v8bf*)(Qrow + c * 32), *(const v8bf*)(Qrow + c * 32 + 16));

  // ---- scores: 36 kv-tiles of 16, accumulator initialized with mask tile ----
  const __bf16* Kbase = Kb + (size_t)(b * LKV) * HID + h * DH + hi * 8;
  const float* Mrow = mask + (size_t)(b * LQ + q0 + hi * 8) * LKV + ln;
  for (int kv0 = 0; kv0 < LKV; kv0 += 16) {
    v8f c;
#pragma unroll
    for (int r = 0; r < 8; ++r) c[r] = Mrow[(size_t)r * LKV + kv0];
    const __bf16* kp = Kbase + (size_t)(kv0 + ln) * HID;
    v16bf kfrag[4];
#pragma unroll
    for (int cc = 0; cc < 4; ++cc)
      kfrag[cc] = cat16(*(const v8bf*)(kp + cc * 32), *(const v8bf*)(kp + cc * 32 + 16));
#pragma unroll
    for (int cc = 0; cc < 4; ++cc)
      c = __builtin_amdgcn_wmma_f32_16x16x32_bf16(false, qa[cc], false, kfrag[cc],
                                                  (short)0, c, false, false);
#pragma unroll
    for (int r = 0; r < 8; ++r)
      Sbuf[(r + hi * 8) * LKV + kv0 + ln] = c[r];
  }

  // ---- full-row softmax (lane-parallel over 576, shfl_xor reductions) ----
  for (int m = 0; m < 16; ++m) {
    float mx = -3.0e38f;
#pragma unroll
    for (int i = 0; i < 18; ++i) mx = fmaxf(mx, Sbuf[m * LKV + lane + i * 32]);
#pragma unroll
    for (int off = 16; off > 0; off >>= 1) mx = fmaxf(mx, __shfl_xor(mx, off, 32));
    float sum = 0.f;
#pragma unroll
    for (int i = 0; i < 18; ++i) {
      float e = __expf(Sbuf[m * LKV + lane + i * 32] - mx);
      sum += e;
      Pbuf[m * LKV + lane + i * 32] = to_bf16(e);
    }
#pragma unroll
    for (int off = 16; off > 0; off >>= 1) sum += __shfl_xor(sum, off, 32);
    if (lane == 0) sums[m] = 1.0f / sum;
  }

  // ---- O = P @ Vh : 18 K-steps of 32 over kv, 8 d-subtiles of 16 ----
  v8f acc[8] = {};
  const __bf16* Vbase = Vt + (size_t)(b * HID + h * DH + ln) * LKV + hi * 8;
  for (int kv0 = 0; kv0 < LKV; kv0 += 32) {
    const __bf16* pp = Pbuf + ln * LKV + kv0 + hi * 8;
    v16bf pa = cat16(*(const v8bf*)pp, *(const v8bf*)(pp + 16));
    v16bf vfrag[8];
#pragma unroll
    for (int j = 0; j < 8; ++j) {
      const __bf16* vp = Vbase + (size_t)(j * 16) * LKV + kv0;
      vfrag[j] = cat16(*(const v8bf*)vp, *(const v8bf*)(vp + 16));
    }
#pragma unroll
    for (int j = 0; j < 8; ++j)
      acc[j] = __builtin_amdgcn_wmma_f32_16x16x32_bf16(false, pa, false, vfrag[j],
                                                       (short)0, acc[j], false, false);
  }

  // ---- epilogue: normalize by 1/rowsum, store bf16 [b*LQ+q, h*128+d] ----
  __bf16* Orow = Attn + (size_t)(b * LQ + q0 + hi * 8) * HID + h * DH + ln;
#pragma unroll
  for (int j = 0; j < 8; ++j) {
#pragma unroll
    for (int r = 0; r < 8; ++r) {
      float v = acc[j][r] * sums[r + hi * 8];
      Orow[(size_t)r * HID + j * 16] = to_bf16(v);
    }
  }
}

// ---------------------------------------------------------------------------
// Host-side orchestration
// ---------------------------------------------------------------------------
extern "C" void kernel_launch(void* const* d_in, const int* in_sizes, int n_in,
                              void* d_out, int out_size, void* d_ws, size_t ws_size,
                              hipStream_t stream) {
  (void)in_sizes; (void)n_in; (void)out_size; (void)ws_size;

  const float* hs   = (const float*)d_in[0];
  const float* img  = (const float*)d_in[1];
  const float* mask = (const float*)d_in[2];
  const float* Wq   = (const float*)d_in[3];
  const float* Wk   = (const float*)d_in[4];
  const float* Wv   = (const float*)d_in[5];
  const float* Wo   = (const float*)d_in[6];

  constexpr size_t SZ_HS  = (size_t)BATCH * LQ * HID;   // 16.8M elems
  constexpr size_t SZ_IMG = (size_t)BATCH * LKV * HID;  // 4.7M elems
  constexpr size_t SZ_W   = (size_t)HID * HID;          // 16.8M elems

  __bf16* hsb  = (__bf16*)d_ws;
  __bf16* imgb = hsb + SZ_HS;
  __bf16* wqb  = imgb + SZ_IMG;
  __bf16* wkb  = wqb + SZ_W;
  __bf16* wvb  = wkb + SZ_W;
  __bf16* wob  = wvb + SZ_W;
  __bf16* qb   = wob + SZ_W;    // Q (pre-scaled by 1/sqrt(D))
  __bf16* kbuf = qb + SZ_HS;    // K
  __bf16* vtb  = kbuf + SZ_IMG; // V transposed per head [B, HID, LKV]
  __bf16* atb  = vtb + SZ_IMG;  // attention output (pre-Wo)

  auto cvt = [&](const float* s, __bf16* d, size_t n) {
    cvt_f32_bf16<<<dim3((unsigned)(n / 4 / 256)), dim3(256), 0, stream>>>(s, d, (long long)n);
  };
  cvt(hs, hsb, SZ_HS);
  cvt(img, imgb, SZ_IMG);
  cvt(Wq, wqb, SZ_W);
  cvt(Wk, wkb, SZ_W);
  cvt(Wv, wvb, SZ_W);
  cvt(Wo, wob, SZ_W);

  const float qscale = 0.08838834764831845f;  // 1/sqrt(128) folded into Q

  // Q = HS @ Wq^T  (bf16 out, scaled)
  gemm_bf16_nt<<<dim3(HID / 128, (BATCH * LQ) / 128), dim3(256), 0, stream>>>(
      hsb, wqb, qb, BATCH * LQ, HID, HID, 0, qscale, 1);
  // K = IMG @ Wk^T
  gemm_bf16_nt<<<dim3(HID / 128, (BATCH * LKV) / 128), dim3(256), 0, stream>>>(
      imgb, wkb, kbuf, BATCH * LKV, HID, HID, 0, 1.0f, 1);
  // Vt = per-batch transpose of IMG @ Wv^T
  gemm_bf16_nt<<<dim3(HID / 128, (BATCH * LKV) / 128), dim3(256), 0, stream>>>(
      imgb, wvb, vtb, BATCH * LKV, HID, HID, 1, 1.0f, LKV);

  // attention: grid over (q-tile, head, batch); 221,440 B dynamic LDS per WG
  const size_t smem = 4 * (16 * LKV * 4) + 4 * (16 * LKV * 2) + 4 * 16 * 4;
  attn_fwd<<<dim3(LQ / 64, NH, BATCH), dim3(128), smem, stream>>>(
      qb, kbuf, vtb, mask, atb);

  // out = Attn @ Wo^T  (f32 out)
  gemm_bf16_nt<<<dim3(HID / 128, (BATCH * LQ) / 128), dim3(256), 0, stream>>>(
      atb, wob, d_out, BATCH * LQ, HID, HID, 2, 1.0f, 1);
}
